// Transolver_52630529245671
// MI455X (gfx1250) — matrix operand order repaired and checked by reading, not compiled, not run on hardware
//
#include <hip/hip_runtime.h>
#include <math.h>

typedef __attribute__((ext_vector_type(16))) _Float16 v16h;
typedef __attribute__((ext_vector_type(8)))  _Float16 v8h;
typedef __attribute__((ext_vector_type(8)))  float    v8f;

union FragH { _Float16 h[16]; v16h v; };

static __device__ __forceinline__ v16h cat8(v8h lo, v8h hi) {
    return __builtin_shufflevector(lo, hi, 0, 1, 2, 3, 4, 5, 6, 7,
                                   8, 9, 10, 11, 12, 13, 14, 15);
}

__device__ __forceinline__ float gelu_f(float x) {
    return 0.5f * x * (1.0f + erff(x * 0.70710678118654752440f));
}

// ---------------------------------------------------------------------------
// concat [x(2) | fx(4) | zeros(26)] -> c32 [M,32]  (K padded to one WMMA step)
// ---------------------------------------------------------------------------
__global__ void concat_kernel(const float* __restrict__ x, const float* __restrict__ fx,
                              float* __restrict__ c32, int M) {
    int m = blockIdx.x * blockDim.x + threadIdx.x;
    if (m >= M) return;
    float* o = c32 + (size_t)m * 32;
    o[0] = x[(size_t)m * 2 + 0];
    o[1] = x[(size_t)m * 2 + 1];
    o[2] = fx[(size_t)m * 4 + 0];
    o[3] = fx[(size_t)m * 4 + 1];
    o[4] = fx[(size_t)m * 4 + 2];
    o[5] = fx[(size_t)m * 4 + 3];
    #pragma unroll
    for (int k = 6; k < 32; ++k) o[k] = 0.0f;
}

// pad pre_w1 [6,512] -> [32,512] (zero rows 6..31)
__global__ void padw1_kernel(const float* __restrict__ w, float* __restrict__ wp) {
    int idx = blockIdx.x * blockDim.x + threadIdx.x;
    if (idx >= 32 * 512) return;
    int r = idx >> 9;
    wp[idx] = (r < 6) ? w[idx] : 0.0f;
}

// ---------------------------------------------------------------------------
// Generic GEMM: C = act(A[M,K] @ W[K,N] + bias (+bias2)) (+ res)
// REQUIRES M%128==0, N%64==0, K%32==0. Epilogue variants are compile-time.
// Block: 256 threads (8 waves), tile 128(M) x 64(N), 32x32 per wave.
// Single-stage software pipeline: prefetch next K-tile during WMMA.
// ---------------------------------------------------------------------------
template <bool HAS_B2, bool ACT, bool HAS_RES>
__global__ __launch_bounds__(256) void gemm_wmma(
    const float* __restrict__ A, const float* __restrict__ W,
    const float* __restrict__ bias, const float* __restrict__ bias2,
    const float* __restrict__ res, float* __restrict__ C,
    int M, int K, int N)
{
    __shared__ alignas(16) _Float16 As[128][40];  // [m][k], +8 pad
    __shared__ alignas(16) _Float16 Bt[64][40];   // [n][k] transposed, +8 pad

    const int tid  = threadIdx.x;
    const int lane = tid & 31;
    const int wid  = tid >> 5;
    const int wm   = wid >> 1;      // 0..3
    const int wn   = wid & 1;       // 0..1
    const int m0   = blockIdx.y * 128;
    const int n0   = blockIdx.x * 64;
    const int h2   = lane >> 4;
    const int lr   = lane & 15;

    // A staging: thread -> row (tid>>1), 16 contiguous k ((tid&1)*16)
    const int ar = tid >> 1;
    const int ac = (tid & 1) * 16;
    // B staging: thread -> k-row (tid>>3), 8 contiguous n ((tid&7)*8)
    const int br = tid >> 3;
    const int bc = (tid & 7) * 8;

    const float* aptr = A + (size_t)(m0 + ar) * K + ac;
    const float* wptr = W + (size_t)br * N + n0 + bc;
    const size_t wstep = (size_t)32 * N;

    v8f acc[2][2];
    #pragma unroll
    for (int i = 0; i < 2; ++i)
        #pragma unroll
        for (int j = 0; j < 2; ++j) acc[i][j] = (v8f){};

    // prologue: fetch k-tile 0 into registers
    float4 a0, a1, a2, a3, w0, w1;
    {
        const float4* ap = (const float4*)aptr;
        a0 = ap[0]; a1 = ap[1]; a2 = ap[2]; a3 = ap[3];
        const float4* wp = (const float4*)wptr;
        w0 = wp[0]; w1 = wp[1];
    }

    for (int kt = 0; kt < K; kt += 32) {
        // ---- stage registers -> LDS ----
        {
            union { v8h v; _Float16 h[8]; } p0, p1;
            p0.h[0] = (_Float16)a0.x; p0.h[1] = (_Float16)a0.y;
            p0.h[2] = (_Float16)a0.z; p0.h[3] = (_Float16)a0.w;
            p0.h[4] = (_Float16)a1.x; p0.h[5] = (_Float16)a1.y;
            p0.h[6] = (_Float16)a1.z; p0.h[7] = (_Float16)a1.w;
            p1.h[0] = (_Float16)a2.x; p1.h[1] = (_Float16)a2.y;
            p1.h[2] = (_Float16)a2.z; p1.h[3] = (_Float16)a2.w;
            p1.h[4] = (_Float16)a3.x; p1.h[5] = (_Float16)a3.y;
            p1.h[6] = (_Float16)a3.z; p1.h[7] = (_Float16)a3.w;
            *(v8h*)&As[ar][ac]     = p0.v;
            *(v8h*)&As[ar][ac + 8] = p1.v;
            Bt[bc + 0][br] = (_Float16)w0.x;
            Bt[bc + 1][br] = (_Float16)w0.y;
            Bt[bc + 2][br] = (_Float16)w0.z;
            Bt[bc + 3][br] = (_Float16)w0.w;
            Bt[bc + 4][br] = (_Float16)w1.x;
            Bt[bc + 5][br] = (_Float16)w1.y;
            Bt[bc + 6][br] = (_Float16)w1.z;
            Bt[bc + 7][br] = (_Float16)w1.w;
        }
        __syncthreads();

        // ---- prefetch next k-tile while WMMAs run ----
        if (kt + 32 < K) {
            const float4* ap = (const float4*)(aptr + kt + 32);
            a0 = ap[0]; a1 = ap[1]; a2 = ap[2]; a3 = ap[3];
            const float4* wp = (const float4*)(wptr + (size_t)(kt + 32) * N);
            w0 = wp[0]; w1 = wp[1];
        }

        // ---- fragments: contiguous 16B LDS loads per half ----
        FragH a[2], b[2];
        #pragma unroll
        for (int i = 0; i < 2; ++i) {
            int row = wm * 32 + i * 16 + lr;
            v8h lo = *(const v8h*)&As[row][8 * h2];        // e=0..7  : k=8h+e
            v8h hi = *(const v8h*)&As[row][8 * h2 + 16];   // e=8..15 : k=8h+e+8
            a[i].v = cat8(lo, hi);
        }
        #pragma unroll
        for (int j = 0; j < 2; ++j) {
            int col = wn * 32 + j * 16 + lr;
            v8h lo = *(const v8h*)&Bt[col][8 * h2];        // e=0..7  : k=8h+e
            v8h hi = *(const v8h*)&Bt[col][8 * h2 + 16];
            b[j].v = cat8(lo, hi);
        }
        #pragma unroll
        for (int i = 0; i < 2; ++i)
            #pragma unroll
            for (int j = 0; j < 2; ++j)
                acc[i][j] = __builtin_amdgcn_wmma_f32_16x16x32_f16(
                    false, a[i].v, false, b[j].v, (short)0, acc[i][j], false, false);
        __syncthreads();
    }

    // ---- straight-line epilogue (variant chosen at compile time) ----
    #pragma unroll
    for (int j = 0; j < 2; ++j) {
        int gc = n0 + wn * 32 + j * 16 + lr;
        float bcol = bias[gc];
        if (HAS_B2) bcol += bias2[gc];
        #pragma unroll
        for (int i = 0; i < 2; ++i) {
            #pragma unroll
            for (int vv = 0; vv < 8; ++vv) {
                int gr = m0 + wm * 32 + i * 16 + vv + 8 * h2;
                float f = acc[i][j][vv] + bcol;
                if (ACT)     f = gelu_f(f);
                if (HAS_RES) f += res[(size_t)gr * N + gc];
                C[(size_t)gr * N + gc] = f;
            }
        }
    }
}

// ---------------------------------------------------------------------------
// LayerNorm over D=256: one block (8 waves) per row.
// ---------------------------------------------------------------------------
__global__ __launch_bounds__(256) void ln_kernel(const float* __restrict__ x,
                                                 const float* __restrict__ g,
                                                 const float* __restrict__ b,
                                                 float* __restrict__ y) {
    int row = blockIdx.x;
    int tid = threadIdx.x;
    float v = x[(size_t)row * 256 + tid];
    float s = v, q = v * v;
    #pragma unroll
    for (int off = 16; off > 0; off >>= 1) {
        s += __shfl_xor(s, off, 32);
        q += __shfl_xor(q, off, 32);
    }
    __shared__ float ss[8], qq[8];
    int wid = tid >> 5, lane = tid & 31;
    if (lane == 0) { ss[wid] = s; qq[wid] = q; }
    __syncthreads();
    if (tid == 0) {
        float S = 0.f, Q = 0.f;
        for (int i = 0; i < 8; ++i) { S += ss[i]; Q += qq[i]; }
        ss[0] = S; qq[0] = Q;
    }
    __syncthreads();
    float mean = ss[0] * (1.0f / 256.0f);
    float var  = qq[0] * (1.0f / 256.0f) - mean * mean;
    float r    = rsqrtf(var + 1e-5f);
    y[(size_t)row * 256 + tid] = (v - mean) * r * g[tid] + b[tid];
}

__global__ void zero_kernel(float* __restrict__ p, int n) {
    int i = blockIdx.x * blockDim.x + threadIdx.x;
    if (i < n) p[i] = 0.0f;
}

// ---------------------------------------------------------------------------
// Temperature softmax over G=64 per (b,n,h) row; one wave per row.
// Also accumulates norm[b,h,g] = sum_n sw via f32 atomics.
// ---------------------------------------------------------------------------
__global__ __launch_bounds__(256) void slice_softmax_kernel(float* __restrict__ sw,
                                                            const float* __restrict__ temp,
                                                            float* __restrict__ norm,
                                                            int NH, int H) {
    int row  = blockIdx.x * 8 + (threadIdx.x >> 5);
    int lane = threadIdx.x & 31;
    int h = row % H;
    int b = row / NH;
    float t = temp[h];
    float* p = sw + (size_t)row * 64;
    float l0 = p[lane]      / t;
    float l1 = p[lane + 32] / t;
    float mx = fmaxf(l0, l1);
    #pragma unroll
    for (int off = 16; off > 0; off >>= 1) mx = fmaxf(mx, __shfl_xor(mx, off, 32));
    float e0 = __expf(l0 - mx), e1 = __expf(l1 - mx);
    float s = e0 + e1;
    #pragma unroll
    for (int off = 16; off > 0; off >>= 1) s += __shfl_xor(s, off, 32);
    float inv = 1.0f / s;
    e0 *= inv; e1 *= inv;
    p[lane] = e0; p[lane + 32] = e1;
    float* np = norm + ((size_t)b * H + h) * 64;
    atomicAdd(&np[lane], e0);
    atomicAdd(&np[lane + 32], e1);
}

// ---------------------------------------------------------------------------
// tok[b,h,g,d] = sum_n sw[b,n,h,g] * fx[b,n,h*32+d]  (split-K over N, atomics)
// grid (B*H, NSPLIT); 8 waves: 4 g-tiles x 2 d-tiles.
// ---------------------------------------------------------------------------
__global__ __launch_bounds__(256) void slice_tok_kernel(const float* __restrict__ sw,
                                                        const float* __restrict__ fx,
                                                        float* __restrict__ tok,
                                                        int N, int H, int KCHUNK) {
    int bh = blockIdx.x;
    int b = bh / H, hh = bh % H;
    int kbase = blockIdx.y * KCHUNK;
    int lane = threadIdx.x & 31, wid = threadIdx.x >> 5;
    int gt = wid >> 1, dt = wid & 1;
    int h2 = lane >> 4, lr = lane & 15;

    v8f acc = (v8f){};
    for (int kk = 0; kk < KCHUNK; kk += 32) {
        FragH a, bf;
        int g = gt * 16 + lr;
        #pragma unroll
        for (int e = 0; e < 16; ++e) {
            int n = kbase + kk + 8 * h2 + (e < 8 ? e : e + 8);
            a.h[e] = (_Float16)sw[(((size_t)b * N + n) * H + hh) * 64 + g];
        }
        int d = dt * 16 + lr;
        #pragma unroll
        for (int e = 0; e < 16; ++e) {
            int n = kbase + kk + 8 * h2 + (e < 8 ? e : e + 8);
            bf.h[e] = (_Float16)fx[((size_t)b * N + n) * 256 + hh * 32 + d];
        }
        acc = __builtin_amdgcn_wmma_f32_16x16x32_f16(false, a.v, false, bf.v,
                                                     (short)0, acc, false, false);
    }
    #pragma unroll
    for (int vv = 0; vv < 8; ++vv) {
        int g = gt * 16 + vv + 8 * h2;
        int d = dt * 16 + lr;
        atomicAdd(&tok[((size_t)bh * 64 + g) * 32 + d], acc[vv]);
    }
}

__global__ void tok_div_kernel(float* __restrict__ tok, const float* __restrict__ norm, int n) {
    int i = blockIdx.x * blockDim.x + threadIdx.x;
    if (i < n) tok[i] = tok[i] / (norm[i >> 5] + 1e-5f);
}

// ---------------------------------------------------------------------------
// Fused q/k/v projection + attention over G=64 tokens per (b,h).
// 64 threads per block, thread g owns token row g.
// ---------------------------------------------------------------------------
__global__ __launch_bounds__(64) void attn_kernel(const float* __restrict__ tok,
                                                  const float* __restrict__ qw,
                                                  const float* __restrict__ kw,
                                                  const float* __restrict__ vw,
                                                  float* __restrict__ os) {
    int bh = blockIdx.x;
    int g  = threadIdx.x;
    __shared__ float ts[64][32];
    __shared__ float ks[64][32], vs[64][32];
    __shared__ float w[3][32][32];
    const float* tb = tok + (size_t)bh * 2048;
    for (int d = 0; d < 32; ++d) ts[g][d] = tb[g * 32 + d];
    for (int idx = g; idx < 1024; idx += 64) {
        int r = idx >> 5, c = idx & 31;
        w[0][r][c] = qw[idx];
        w[1][r][c] = kw[idx];
        w[2][r][c] = vw[idx];
    }
    __syncthreads();
    float qr[32];
    for (int d = 0; d < 32; ++d) {
        float aq = 0.f, ak = 0.f, av = 0.f;
        for (int e = 0; e < 32; ++e) {
            float t = ts[g][e];
            aq += t * w[0][e][d];
            ak += t * w[1][e][d];
            av += t * w[2][e][d];
        }
        qr[d] = aq; ks[g][d] = ak; vs[g][d] = av;
    }
    __syncthreads();
    const float scale = 0.17677669529663687f;  // 1/sqrt(32)
    float s[64];
    float mx = -1e30f;
    for (int m = 0; m < 64; ++m) {
        float a = 0.f;
        #pragma unroll
        for (int d = 0; d < 32; ++d) a += qr[d] * ks[m][d];
        a *= scale; s[m] = a; mx = fmaxf(mx, a);
    }
    float sum = 0.f;
    for (int m = 0; m < 64; ++m) { s[m] = __expf(s[m] - mx); sum += s[m]; }
    float inv = 1.0f / sum;
    float* ob = os + (size_t)bh * 2048 + g * 32;
    for (int d = 0; d < 32; ++d) {
        float a = 0.f;
        for (int m = 0; m < 64; ++m) a += s[m] * vs[m][d];
        ob[d] = a * inv;
    }
}

// ---------------------------------------------------------------------------
// out_x[b,n,h*32+d] = sum_g os[b,h,g,d] * sw[b,n,h,g]
// grid (B*H, N/128); 8 waves x 16 rows; K=64 (2 WMMA steps), 2 d-tiles.
// ---------------------------------------------------------------------------
__global__ __launch_bounds__(256) void deslice_kernel(const float* __restrict__ sw,
                                                      const float* __restrict__ os,
                                                      float* __restrict__ xo,
                                                      int N, int H) {
    int bh = blockIdx.x;
    int b = bh / H, hh = bh % H;
    int nbase = blockIdx.y * 128;
    int lane = threadIdx.x & 31, wid = threadIdx.x >> 5;
    int h2 = lane >> 4, lr = lane & 15;
    int row = nbase + wid * 16 + lr;

    v8f acc[2];
    acc[0] = (v8f){}; acc[1] = (v8f){};
    for (int kk = 0; kk < 64; kk += 32) {
        FragH a;
        #pragma unroll
        for (int e = 0; e < 16; ++e) {
            int g = kk + 8 * h2 + (e < 8 ? e : e + 8);
            a.h[e] = (_Float16)sw[(((size_t)b * N + row) * H + hh) * 64 + g];
        }
        #pragma unroll
        for (int j = 0; j < 2; ++j) {
            FragH bf;
            int d = j * 16 + lr;
            #pragma unroll
            for (int e = 0; e < 16; ++e) {
                int g = kk + 8 * h2 + (e < 8 ? e : e + 8);
                bf.h[e] = (_Float16)os[((size_t)bh * 64 + g) * 32 + d];
            }
            acc[j] = __builtin_amdgcn_wmma_f32_16x16x32_f16(false, a.v, false, bf.v,
                                                            (short)0, acc[j], false, false);
        }
    }
    #pragma unroll
    for (int j = 0; j < 2; ++j)
        #pragma unroll
        for (int vv = 0; vv < 8; ++vv) {
            int n = nbase + wid * 16 + vv + 8 * h2;
            int d = j * 16 + lr;
            xo[((size_t)b * N + n) * 256 + hh * 32 + d] = acc[j][vv];
        }
}

// ---------------------------------------------------------------------------
// Head output GEMM: [M,256] @ [256,4] + b  (N too small for WMMA tiles)
// ---------------------------------------------------------------------------
__global__ void head2_kernel(const float* __restrict__ A, const float* __restrict__ W,
                             const float* __restrict__ b, float* __restrict__ out, int M) {
    int idx = blockIdx.x * blockDim.x + threadIdx.x;
    if (idx >= M * 4) return;
    int m = idx >> 2, o = idx & 3;
    const float* ar = A + (size_t)m * 256;
    float acc = b[o];
    for (int k = 0; k < 256; ++k) acc += ar[k] * W[k * 4 + o];
    out[idx] = acc;
}

// ---------------------------------------------------------------------------
extern "C" void kernel_launch(void* const* d_in, const int* in_sizes, int n_in,
                              void* d_out, int out_size, void* d_ws, size_t ws_size,
                              hipStream_t stream) {
    (void)in_sizes; (void)n_in; (void)out_size; (void)ws_size;
    const float* x        = (const float*)d_in[0];
    const float* fx_in    = (const float*)d_in[1];
    const float* pre_w1   = (const float*)d_in[2];
    const float* pre_b1   = (const float*)d_in[3];
    const float* pre_w2   = (const float*)d_in[4];
    const float* pre_b2   = (const float*)d_in[5];
    const float* placeholder = (const float*)d_in[6];
    const float* ln1_g    = (const float*)d_in[7];
    const float* ln1_b    = (const float*)d_in[8];
    const float* px_w     = (const float*)d_in[9];
    const float* px_b     = (const float*)d_in[10];
    const float* pfx_w    = (const float*)d_in[11];
    const float* pfx_b    = (const float*)d_in[12];
    const float* slice_w  = (const float*)d_in[13];
    const float* slice_b  = (const float*)d_in[14];
    const float* temp     = (const float*)d_in[15];
    const float* q_w      = (const float*)d_in[16];
    const float* k_w      = (const float*)d_in[17];
    const float* v_w      = (const float*)d_in[18];
    const float* out_w    = (const float*)d_in[19];
    const float* out_b    = (const float*)d_in[20];
    const float* ln2_g    = (const float*)d_in[21];
    const float* ln2_b    = (const float*)d_in[22];
    const float* mlp_w1   = (const float*)d_in[23];
    const float* mlp_b1   = (const float*)d_in[24];
    const float* mlp_w2   = (const float*)d_in[25];
    const float* mlp_b2   = (const float*)d_in[26];
    const float* ln3_g    = (const float*)d_in[27];
    const float* ln3_b    = (const float*)d_in[28];
    const float* head_w1  = (const float*)d_in[29];
    const float* head_b1  = (const float*)d_in[30];
    const float* head_w2  = (const float*)d_in[31];
    const float* head_b2  = (const float*)d_in[32];

    const int B = 2, N = 16384, D = 256, H = 8, dh = 32, G = 64, L = 5;
    const int M = B * N;  // 32768

    float* ws = (float*)d_ws;
    size_t off = 0;
    float* hbuf  = ws + off; off += (size_t)M * D;          // residual stream
    float* ybuf  = ws + off; off += (size_t)M * D;          // LN output
    float* abuf  = ws + off; off += (size_t)M * D;          // x_mid / mlp hidden / head hidden
    float* fbuf  = ws + off; off += (size_t)M * D;          // fx_mid / out_x
    float* swb   = ws + off; off += (size_t)M * H * G;      // pre1 hidden & sw (16.7M)
    float* normb = ws + off; off += (size_t)B * H * G;      // 1024
    float* tokb  = ws + off; off += (size_t)B * H * G * dh; // 32768 (contiguous after normb)
    float* osb   = ws + off; off += (size_t)B * H * G * dh;
    float* c32   = ws + off; off += (size_t)M * 32;         // padded concat input
    float* w1pad = ws + off; off += (size_t)32 * 512;       // padded pre_w1

    // preprocess: h = gelu([x|fx|0] @ pre_w1_pad + b1) @ pre_w2 + b2 + placeholder
    concat_kernel<<<(M + 255) / 256, 256, 0, stream>>>(x, fx_in, c32, M);
    padw1_kernel<<<(32 * 512 + 255) / 256, 256, 0, stream>>>(pre_w1, w1pad);
    gemm_wmma<false, true, false><<<dim3(512 / 64, M / 128), 256, 0, stream>>>(
        c32, w1pad, pre_b1, nullptr, nullptr, swb, M, 32, 512);
    gemm_wmma<true, false, false><<<dim3(256 / 64, M / 128), 256, 0, stream>>>(
        swb, pre_w2, pre_b2, placeholder, nullptr, hbuf, M, 512, 256);

    for (int i = 0; i < L; ++i) {
        ln_kernel<<<M, 256, 0, stream>>>(hbuf, ln1_g + i * D, ln1_b + i * D, ybuf);
        gemm_wmma<false, false, false><<<dim3(4, M / 128), 256, 0, stream>>>(
            ybuf, pfx_w + (size_t)i * D * D, pfx_b + i * D, nullptr, nullptr, fbuf, M, 256, 256);
        gemm_wmma<false, false, false><<<dim3(4, M / 128), 256, 0, stream>>>(
            ybuf, px_w + (size_t)i * D * D, px_b + i * D, nullptr, nullptr, abuf, M, 256, 256);
        // slice logits: [M*H, 32] @ [32, 64]
        gemm_wmma<false, false, false><<<dim3(1, (M * H) / 128), 256, 0, stream>>>(
            abuf, slice_w + (size_t)i * dh * G, slice_b + i * G, nullptr, nullptr, swb,
            M * H, dh, G);
        zero_kernel<<<(B * H * G + B * H * G * dh + 255) / 256, 256, 0, stream>>>(
            normb, B * H * G + B * H * G * dh);
        slice_softmax_kernel<<<(M * H) / 8, 256, 0, stream>>>(swb, temp + i * H, normb, N * H, H);
        slice_tok_kernel<<<dim3(B * H, 64), 256, 0, stream>>>(swb, fbuf, tokb, N, H, N / 64);
        tok_div_kernel<<<(B * H * G * dh + 255) / 256, 256, 0, stream>>>(
            tokb, normb, B * H * G * dh);
        // fused qkv + attention over G=64 tokens
        attn_kernel<<<B * H, 64, 0, stream>>>(tokb, q_w + (size_t)i * dh * dh,
                                              k_w + (size_t)i * dh * dh,
                                              v_w + (size_t)i * dh * dh, osb);
        deslice_kernel<<<dim3(B * H, N / 128), 256, 0, stream>>>(swb, osb, fbuf, N, H);
        // h = h + out_x @ out_w + out_b
        gemm_wmma<false, false, true><<<dim3(4, M / 128), 256, 0, stream>>>(
            fbuf, out_w + (size_t)i * D * D, out_b + i * D, nullptr, hbuf, hbuf, M, 256, 256);
        // MLP
        ln_kernel<<<M, 256, 0, stream>>>(hbuf, ln2_g + i * D, ln2_b + i * D, ybuf);
        gemm_wmma<false, true, false><<<dim3(4, M / 128), 256, 0, stream>>>(
            ybuf, mlp_w1 + (size_t)i * D * D, mlp_b1 + i * D, nullptr, nullptr, abuf, M, 256, 256);
        gemm_wmma<false, false, true><<<dim3(4, M / 128), 256, 0, stream>>>(
            abuf, mlp_w2 + (size_t)i * D * D, mlp_b2 + i * D, nullptr, hbuf, hbuf, M, 256, 256);
    }

    ln_kernel<<<M, 256, 0, stream>>>(hbuf, ln3_g, ln3_b, ybuf);
    gemm_wmma<false, true, false><<<dim3(4, M / 128), 256, 0, stream>>>(
        ybuf, head_w1, head_b1, nullptr, nullptr, abuf, M, 256, 256);
    head2_kernel<<<(M * 4 + 255) / 256, 256, 0, stream>>>(
        abuf, head_w2, head_b2, (float*)d_out, M);
}
